// CrossDomainGAT_82197084111151
// MI455X (gfx1250) — matrix operand order repaired and checked by reading, not compiled, not run on hardware
//
#include <hip/hip_runtime.h>
#include <hip/hip_bf16.h>

#define N_NODES 50000
#define N_EDGES 800000
#define DIM     64
#define HIDV    128
#define BATCH   16384

typedef __attribute__((ext_vector_type(16))) _Float16 v16h;
typedef __attribute__((ext_vector_type(8)))  _Float16 v8h;
typedef __attribute__((ext_vector_type(8)))  float    v8f;

// ---------------- order-preserving float<->uint for atomic max ----------------
__device__ __forceinline__ unsigned f2ord(float f) {
    unsigned u = __float_as_uint(f);
    return (u & 0x80000000u) ? ~u : (u | 0x80000000u);
}
__device__ __forceinline__ float ord2f(unsigned o) {
    return (o & 0x80000000u) ? __uint_as_float(o & 0x7fffffffu)
                             : __uint_as_float(~o);
}

// ---------------- generic WMMA GEMM: out[M x NC] = A[M x K](f16) * W[NC x K]^T --
// One wave computes a 16-row stripe across all NC columns; A tile kept in regs.
template<int K, int NC, bool HALF_OUT>
__global__ __launch_bounds__(256)
void gemm16(const _Float16* __restrict__ A, const _Float16* __restrict__ W,
            const float* __restrict__ bias, void* __restrict__ out,
            int ldc, int M)
{
    const int lane   = threadIdx.x & 31;
    const int wave   = threadIdx.x >> 5;
    const int stripe = blockIdx.x * 8 + wave;
    if (stripe * 16 >= M) return;
    const int m0 = stripe * 16;
    const int r  = lane & 15;          // row within A-tile / col within B-tile
    const int hi = lane >> 4;

    constexpr int KT = K / 32;
    constexpr int NT = NC / 16;
    v8f acc[NT];
#pragma unroll
    for (int j = 0; j < NT; ++j) acc[j] = v8f{};

    const _Float16* arow = A + (size_t)(m0 + r) * K;
#pragma unroll
    for (int kt = 0; kt < KT; ++kt) {
        const int k0 = kt * 32;
        // A 16x32 f16 layout: lane r (half L) / r (half H): halves 0..7 -> K=k0+8*hi..,
        // halves 8..15 -> K=k0+16+8*hi..  (two contiguous 16B chunks)
        union { v16h v; v8h h8[2]; } a;
        a.h8[0] = *(const v8h*)(arow + k0 + 8 * hi);
        a.h8[1] = *(const v8h*)(arow + k0 + 16 + 8 * hi);
#pragma unroll
        for (int j = 0; j < NT; ++j) {
            // B 32x16 f16 layout: lane col = r, halves 0..7 -> K = k0+8*hi.., halves 8..15 -> K = k0+16+8*hi..
            const _Float16* wrow = W + (size_t)(j * 16 + r) * K + k0 + 8 * hi;
            union { v16h v; v8h h8[2]; } b;
            b.h8[0] = *(const v8h*)(wrow);
            b.h8[1] = *(const v8h*)(wrow + 16);
            acc[j] = __builtin_amdgcn_wmma_f32_16x16x32_f16(
                false, a.v, false, b.v, (short)0, acc[j], false, false);
        }
    }
    // C/D layout: lane col = r, VGPR v -> row v + 8*hi
#pragma unroll
    for (int j = 0; j < NT; ++j) {
        const int col = j * 16 + r;
        const float bv = bias ? bias[col] : 0.0f;
#pragma unroll
        for (int v = 0; v < 8; ++v) {
            const int row = m0 + v + 8 * hi;
            float val = acc[j][v] + bv;
            if (HALF_OUT)
                ((_Float16*)out)[(size_t)row * ldc + col] = (_Float16)val;
            else
                ((float*)out)[(size_t)row * ldc + col] = val;
        }
    }
}

// ---------------- small helper kernels ----------------------------------------
__global__ void f32_to_f16_k(const float* __restrict__ a, _Float16* __restrict__ o, int n) {
    int i = blockIdx.x * blockDim.x + threadIdx.x;
    if (i < n) o[i] = (_Float16)a[i];
}

__global__ void gather_emb_f16_k(const float* __restrict__ emb, const int* __restrict__ x,
                                 _Float16* __restrict__ o, int n) {
    int i = blockIdx.x * blockDim.x + threadIdx.x;
    if (i >= n) return;
    int node = x[i >> 6];
    o[i] = (_Float16)emb[(size_t)node * DIM + (i & 63)];
}

__global__ void att_dots_k(const float* __restrict__ z, const float* __restrict__ atts,
                           const float* __restrict__ attd, float* __restrict__ as_,
                           float* __restrict__ ad_, int NH, int H, int C) {
    int t = blockIdx.x * blockDim.x + threadIdx.x;
    if (t >= NH) return;
    int h = t % H;
    const float* zr = z + (size_t)t * C;
    float s = 0.f, d = 0.f;
    for (int c = 0; c < C; ++c) { s += zr[c] * atts[h * C + c]; d += zr[c] * attd[h * C + c]; }
    as_[t] = s; ad_[t] = d;
}

template<int H>
__global__ void edge_softmax_pass_k(const int* __restrict__ ei, int E_, int N_,
                                    const float* __restrict__ as_, const float* __restrict__ ad_,
                                    unsigned* __restrict__ amax, float* __restrict__ den,
                                    int pass) {
    int t = blockIdx.x * blockDim.x + threadIdx.x;
    int tot = (E_ + N_) * H;
    if (t >= tot) return;
    int e = t / H, h = t - e * H;
    int s, d;
    if (e < E_) { s = ei[e]; d = ei[E_ + e]; } else { s = d = e - E_; }
    float a = as_[s * H + h] + ad_[d * H + h];
    a = a > 0.f ? a : 0.2f * a;                       // leaky_relu 0.2
    if (pass == 0) {
        atomicMax(amax + d * H + h, f2ord(a));
    } else {
        float m = ord2f(amax[d * H + h]);
        atomicAdd(den + d * H + h, __expf(a - m));
    }
}

template<int H, int C>
__global__ __launch_bounds__(256)
void edge_aggregate_k(const int* __restrict__ ei, int E_, int N_,
                      const float* __restrict__ as_, const float* __restrict__ ad_,
                      const unsigned* __restrict__ amax, const float* __restrict__ den,
                      const float* __restrict__ z, float* __restrict__ agg) {
    constexpr int HID_ = H * C;
    constexpr int VEC  = HID_ / 32;
    int lane = threadIdx.x & 31;
    int e = (blockIdx.x * blockDim.x + threadIdx.x) >> 5;
    if (e >= E_ + N_) return;
    int s, d;
    if (e < E_) { s = ei[e]; d = ei[E_ + e]; } else { s = d = e - E_; }
    int c0 = lane * VEC;
    int h  = c0 / C;
    float a = as_[s * H + h] + ad_[d * H + h];
    a = a > 0.f ? a : 0.2f * a;
    float w = __expf(a - ord2f(amax[d * H + h])) / den[d * H + h];
    const float* zs = z + (size_t)s * HID_ + c0;
    float* od = agg + (size_t)d * HID_ + c0;
#pragma unroll
    for (int i = 0; i < VEC; ++i) atomicAdd(od + i, zs[i] * w);
}

__global__ void finalize_elu_f16_k(const float* __restrict__ agg, const float* __restrict__ bias,
                                   _Float16* __restrict__ o, int n, int HID_) {
    int i = blockIdx.x * blockDim.x + threadIdx.x;
    if (i >= n) return;
    float v = agg[i] + bias[i % HID_];
    v = v > 0.f ? v : (__expf(v) - 1.f);              // ELU
    o[i] = (_Float16)v;
}

__global__ void add_bias_f32_k(const float* __restrict__ agg, const float* __restrict__ bias,
                               float* __restrict__ o, int n, int HID_) {
    int i = blockIdx.x * blockDim.x + threadIdx.x;
    if (i >= n) return;
    o[i] = agg[i] + bias[i % HID_];
}

__global__ void gather_rows_f16_k(const float* __restrict__ src, const int* __restrict__ idx,
                                  _Float16* __restrict__ o, int B_, int Dm) {
    int i = blockIdx.x * blockDim.x + threadIdx.x;
    if (i >= B_ * Dm) return;
    int b = i / Dm, k = i - b * Dm;
    o[i] = (_Float16)src[(size_t)idx[b] * Dm + k];
}

__global__ void make_wfs_k(const float* __restrict__ Wf1, _Float16* __restrict__ o) {
    int i = blockIdx.x * blockDim.x + threadIdx.x;
    if (i >= 128 * 64) return;
    int r = i >> 6, k = i & 63;
    o[i] = (_Float16)(Wf1[r * 128 + k] + Wf1[r * 128 + 64 + k]);
}

__global__ __launch_bounds__(128)
void bn_stats_k(const float* __restrict__ z, float* __restrict__ sum, float* __restrict__ sq,
                int Cn, int rowsPerBlock) {
    int col = threadIdx.x;
    int r0  = blockIdx.x * rowsPerBlock;
    float s = 0.f, q = 0.f;
    for (int r = 0; r < rowsPerBlock; ++r) {
        float v = z[(size_t)(r0 + r) * Cn + col];
        s += v; q += v * v;
    }
    atomicAdd(sum + col, s);
    atomicAdd(sq + col, q);
}

__global__ void bn_apply_relu_f16_k(const float* __restrict__ z, const float* __restrict__ sum,
                                    const float* __restrict__ sq, const float* __restrict__ g,
                                    const float* __restrict__ beta, _Float16* __restrict__ o,
                                    int B_, int Cn) {
    int i = blockIdx.x * blockDim.x + threadIdx.x;
    if (i >= B_ * Cn) return;
    int c = i % Cn;
    float inv = 1.f / (float)B_;
    float mu  = sum[c] * inv;
    float var = sq[c] * inv - mu * mu;                // biased var (training-mode BN)
    float v = (z[i] - mu) * rsqrtf(var + 1e-5f) * g[c] + beta[c];
    o[i] = (_Float16)(v > 0.f ? v : 0.f);
}

__global__ void predictor_final_k(const float* __restrict__ z, const float* __restrict__ w3,
                                  const float* __restrict__ b3, float* __restrict__ out, int B_) {
    int b = blockIdx.x * blockDim.x + threadIdx.x;
    if (b >= B_) return;
    const float* zr = z + (size_t)b * 64;
    float s = b3[0];
#pragma unroll
    for (int k = 0; k < 64; ++k) {
        float v = zr[k];
        v = v > 0.f ? v : 0.f;                        // ReLU fused before final dot
        s += v * w3[k];
    }
    out[b] = s;
}

// ---------------- host orchestration ------------------------------------------
extern "C" void kernel_launch(void* const* d_in, const int* in_sizes, int n_in,
                              void* d_out, int out_size, void* d_ws, size_t ws_size,
                              hipStream_t stream) {
    (void)in_sizes; (void)n_in; (void)out_size; (void)ws_size;
    const float* emb   = (const float*)d_in[0];
    const float* W1    = (const float*)d_in[1];
    const float* atts1 = (const float*)d_in[2];
    const float* attd1 = (const float*)d_in[3];
    const float* b1    = (const float*)d_in[4];
    const float* W2    = (const float*)d_in[5];
    const float* atts2 = (const float*)d_in[6];
    const float* attd2 = (const float*)d_in[7];
    const float* b2    = (const float*)d_in[8];
    const float* Wf1   = (const float*)d_in[9];
    const float* bf1   = (const float*)d_in[10];
    const float* gf    = (const float*)d_in[11];
    const float* betaf = (const float*)d_in[12];
    const float* Wf2   = (const float*)d_in[13];
    const float* bf2   = (const float*)d_in[14];
    const float* Wp1   = (const float*)d_in[15];
    const float* bp1   = (const float*)d_in[16];
    const float* gp    = (const float*)d_in[17];
    const float* betap = (const float*)d_in[18];
    const float* Wp2   = (const float*)d_in[19];
    const float* bp2   = (const float*)d_in[20];
    const float* Wp3   = (const float*)d_in[21];
    const float* bp3   = (const float*)d_in[22];
    const int*   x     = (const int*)d_in[23];
    const int*   ei    = (const int*)d_in[24];
    const int*   uidx  = (const int*)d_in[25];
    const int*   iidx  = (const int*)d_in[26];

    // ---- workspace overlay (~72 MB peak) ----
    char* base = (char*)d_ws;
    size_t off = 0;
    auto take = [&](size_t bytes) -> char* {
        char* p = base + off;
        off += (bytes + 255) & ~(size_t)255;
        return p;
    };
    _Float16* W1h  = (_Float16*)take(128 * 64 * 2);
    _Float16* W2h  = (_Float16*)take(64 * 128 * 2);
    _Float16* Wfsh = (_Float16*)take(128 * 64 * 2);
    _Float16* Wf2h = (_Float16*)take(64 * 128 * 2);
    _Float16* Wp1h = (_Float16*)take(128 * 128 * 2);
    _Float16* Wp2h = (_Float16*)take(64 * 128 * 2);
    float* bnsum = (float*)take(128 * 4);   // bnsq contiguous right after (512B pad)
    float* bnsq  = (float*)take(128 * 4);
    char* slot1 = take((size_t)26 << 20);
    char* slot2 = take((size_t)26 << 20);
    char* slot3 = take((size_t)14 << 20);
    char* slot4 = take((size_t)5  << 20);

    const int Nn = N_NODES, Ee = N_EDGES, Bb = BATCH;
    const int ET = Ee + Nn;

    // slot aliases (lifetimes verified: each reuse happens after producer/consumers done)
    _Float16* h16  = (_Float16*)slot3;                  // [N,64]
    float*    z1   = (float*)slot1;                     // [N,128]
    float*    as1  = (float*)(slot4);
    float*    ad1  = (float*)(slot4 + 0x100000);
    unsigned* amax1= (unsigned*)(slot4 + 0x200000);
    float*    den1 = (float*)(slot4 + 0x300000);
    float*    agg1 = (float*)slot2;                     // [N,128]
    _Float16* h1h  = (_Float16*)slot3;                  // [N,128] (h16 dead)
    float*    z2   = (float*)slot1;                     // [N,64]  (z1 dead)
    float*    as2  = (float*)(slot4);
    float*    ad2  = (float*)(slot4 + 0x100000);
    unsigned* amax2= (unsigned*)(slot4 + 0x200000);
    float*    den2 = (float*)(slot4 + 0x300000);
    float*    agg2 = (float*)slot2;                     // [N,64] (agg1 dead)
    float*    h2   = (float*)slot3;                     // [N,64] (h1h dead)
    _Float16* gu16 = (_Float16*)slot4;                  // [B,64] (att arrays dead)
    _Float16* gi16 = (_Float16*)(slot4 + 0x220000);
    float*    fz   = (float*)slot1;                     // [B,128] (z2 dead)
    _Float16* fzr16= (_Float16*)(slot1 + 0x900000);     // [B,128] f16
    _Float16* cat16= (_Float16*)slot2;                  // [B,128] (agg2 dead)
    float*    pz   = (float*)slot1;                     // [B,128] (fz dead)
    _Float16* pzr16= (_Float16*)(slot1 + 0x900000);
    float*    pz2  = (float*)(slot2 + 0x500000);        // [B,64]

    const int T = 256;
    auto cdiv = [](int a, int b) { return (a + b - 1) / b; };

    // ---- weight conversions ----
    f32_to_f16_k<<<cdiv(8192, T), T, 0, stream>>>(W1,  W1h,  8192);
    f32_to_f16_k<<<cdiv(8192, T), T, 0, stream>>>(W2,  W2h,  8192);
    f32_to_f16_k<<<cdiv(8192, T), T, 0, stream>>>(Wf2, Wf2h, 8192);
    f32_to_f16_k<<<cdiv(16384, T), T, 0, stream>>>(Wp1, Wp1h, 16384);
    f32_to_f16_k<<<cdiv(8192, T), T, 0, stream>>>(Wp2, Wp2h, 8192);
    make_wfs_k<<<cdiv(8192, T), T, 0, stream>>>(Wf1, Wfsh);
    gather_emb_f16_k<<<cdiv(Nn * 64, T), T, 0, stream>>>(emb, x, h16, Nn * 64);

    // ---- GAT layer 1 ----
    gemm16<64, 128, false><<<cdiv(Nn / 16, 8), T, 0, stream>>>(h16, W1h, nullptr, z1, 128, Nn);
    att_dots_k<<<cdiv(Nn * 4, T), T, 0, stream>>>(z1, atts1, attd1, as1, ad1, Nn * 4, 4, 32);
    hipMemsetAsync(amax1, 0, (size_t)Nn * 4 * 4, stream);
    hipMemsetAsync(den1,  0, (size_t)Nn * 4 * 4, stream);
    hipMemsetAsync(agg1,  0, (size_t)Nn * 128 * 4, stream);
    edge_softmax_pass_k<4><<<cdiv(ET * 4, T), T, 0, stream>>>(ei, Ee, Nn, as1, ad1, amax1, den1, 0);
    edge_softmax_pass_k<4><<<cdiv(ET * 4, T), T, 0, stream>>>(ei, Ee, Nn, as1, ad1, amax1, den1, 1);
    edge_aggregate_k<4, 32><<<cdiv(ET, 8), T, 0, stream>>>(ei, Ee, Nn, as1, ad1, amax1, den1, z1, agg1);
    finalize_elu_f16_k<<<cdiv(Nn * 128, T), T, 0, stream>>>(agg1, b1, h1h, Nn * 128, 128);

    // ---- GAT layer 2 ----
    gemm16<128, 64, false><<<cdiv(Nn / 16, 8), T, 0, stream>>>(h1h, W2h, nullptr, z2, 64, Nn);
    att_dots_k<<<cdiv(Nn, T), T, 0, stream>>>(z2, atts2, attd2, as2, ad2, Nn, 1, 64);
    hipMemsetAsync(amax2, 0, (size_t)Nn * 4, stream);
    hipMemsetAsync(den2,  0, (size_t)Nn * 4, stream);
    hipMemsetAsync(agg2,  0, (size_t)Nn * 64 * 4, stream);
    edge_softmax_pass_k<1><<<cdiv(ET, T), T, 0, stream>>>(ei, Ee, Nn, as2, ad2, amax2, den2, 0);
    edge_softmax_pass_k<1><<<cdiv(ET, T), T, 0, stream>>>(ei, Ee, Nn, as2, ad2, amax2, den2, 1);
    edge_aggregate_k<1, 64><<<cdiv(ET, 8), T, 0, stream>>>(ei, Ee, Nn, as2, ad2, amax2, den2, z2, agg2);
    add_bias_f32_k<<<cdiv(Nn * 64, T), T, 0, stream>>>(agg2, b2, h2, Nn * 64, 64);

    // ---- gather user/item embeddings ----
    gather_rows_f16_k<<<cdiv(Bb * 64, T), T, 0, stream>>>(h2, uidx, gu16, Bb, 64);
    gather_rows_f16_k<<<cdiv(Bb * 64, T), T, 0, stream>>>(h2, iidx, gi16, Bb, 64);

    const int gb = cdiv(Bb / 16, 8);   // 128 blocks for B=16384

    // ---- fusion (user) -> cat cols [0,64) ----
    gemm16<64, 128, false><<<gb, T, 0, stream>>>(gu16, Wfsh, bf1, fz, 128, Bb);
    hipMemsetAsync(bnsum, 0, 1024, stream);  // covers bnsum + bnsq (contiguous)
    bn_stats_k<<<128, 128, 0, stream>>>(fz, bnsum, bnsq, 128, 128);
    bn_apply_relu_f16_k<<<cdiv(Bb * 128, T), T, 0, stream>>>(fz, bnsum, bnsq, gf, betaf, fzr16, Bb, 128);
    gemm16<128, 64, true><<<gb, T, 0, stream>>>(fzr16, Wf2h, bf2, cat16, 128, Bb);

    // ---- fusion (item) -> cat cols [64,128) ----
    gemm16<64, 128, false><<<gb, T, 0, stream>>>(gi16, Wfsh, bf1, fz, 128, Bb);
    hipMemsetAsync(bnsum, 0, 1024, stream);
    bn_stats_k<<<128, 128, 0, stream>>>(fz, bnsum, bnsq, 128, 128);
    bn_apply_relu_f16_k<<<cdiv(Bb * 128, T), T, 0, stream>>>(fz, bnsum, bnsq, gf, betaf, fzr16, Bb, 128);
    gemm16<128, 64, true><<<gb, T, 0, stream>>>(fzr16, Wf2h, bf2, cat16 + 64, 128, Bb);

    // ---- predictor ----
    gemm16<128, 128, false><<<gb, T, 0, stream>>>(cat16, Wp1h, bp1, pz, 128, Bb);
    hipMemsetAsync(bnsum, 0, 1024, stream);
    bn_stats_k<<<128, 128, 0, stream>>>(pz, bnsum, bnsq, 128, 128);
    bn_apply_relu_f16_k<<<cdiv(Bb * 128, T), T, 0, stream>>>(pz, bnsum, bnsq, gp, betap, pzr16, Bb, 128);
    gemm16<128, 64, false><<<gb, T, 0, stream>>>(pzr16, Wp2h, bp2, pz2, 64, Bb);
    predictor_final_k<<<cdiv(Bb, T), T, 0, stream>>>(pz2, Wp3, bp3, (float*)d_out, Bb);
}
